// PRNN_68942815035711
// MI455X (gfx1250) — matrix operand order repaired and checked by reading, not compiled, not run on hardware
//
#include <hip/hip_runtime.h>

// ---------------------------------------------------------------------------
// PRNN: GRU(128) x 32 steps + MLP head, fused, f16 WMMA with f32 accumulate.
// B=16384, L=64, P=2, T=32, NH=128, NOUT=4.
// Each wave32 owns a 16-row batch tile; 8 waves/block (2 per SIMD32) so one
// wave's v_wmma issue hides the other's ds_load latency. Weights LDS-resident
// in f16; x tile staged by the Tensor Data Mover (async, TENSORcnt); gate
// nonlinearities use the CDNA5 hardware TANH transcendental.
// ---------------------------------------------------------------------------

#define NH      128
#define T_STEPS 32
#define NGATE   384
#define HS      136            // padded LDS row stride (halves): 272B = 17*16B, bank-spread
#define WAVES   8
#define ROWS_PER_BLOCK (WAVES * 16)    // 128
#define BATCH   16384

typedef __attribute__((ext_vector_type(16))) _Float16 v16h;
typedef __attribute__((ext_vector_type(8)))  _Float16 v8h;
typedef __attribute__((ext_vector_type(8)))  float    v8f;
typedef __attribute__((ext_vector_type(4)))  unsigned u32x4;
typedef __attribute__((ext_vector_type(8)))  int      i32x8;
typedef __attribute__((ext_vector_type(4)))  int      i32x4;

// LDS budget (bytes) -- must match the carve in the kernel
#define WHH_HALVES (NGATE * HS)        // 52224
#define W1_HALVES  (NH * HS)           // 17408
#define W2_HALVES  (16 * HS)           // 2176
#define LUT_FLOATS (4 * NGATE)         // 1536
#define HBUF_HALVES (WAVES * 16 * HS)  // 17408 halves = 34816 B (x stage 32 KB fits)
#define SMEM_BYTES ((WHH_HALVES + W1_HALVES + W2_HALVES) * 2 \
                  + (LUT_FLOATS + NH + NH + 16) * 4          \
                  + (ROWS_PER_BLOCK * 2) * 4                 \
                  + 3 * HBUF_HALVES * 2)                     // = 256320 < 320 KB

// ---- WMMA wrapper: D = A(16x32 f16) x B(32x16 f16) + C(16x16 f32) ----------
__device__ __forceinline__ v8f wmma_f16(v16h a, v16h b, v8f c) {
  return __builtin_amdgcn_wmma_f32_16x16x32_f16(false, a, false, b,
                                                (short)0, c, false, false);
}

// A-operand fragment (16x32 f16) from a row-major LDS matrix.
// lanes 0-15 row M=lane: VGPR0-3 K=k0..k0+7, VGPR4-7 K=k0+16..23;
// lanes 16-31 same rows: K=k0+8..15 / k0+24..31.  Two contiguous b128 loads.
__device__ __forceinline__ v16h load_frag_A(const _Float16* base, int stride,
                                            int k0, int lane) {
  int m   = lane & 15;
  int off = (lane >> 4) << 3;          // 0 or 8
  const _Float16* p = base + m * stride + k0 + off;
  v8h lo = *(const v8h*)(p);
  v8h hi = *(const v8h*)(p + 16);
  return __builtin_shufflevector(lo, hi, 0,1,2,3,4,5,6,7,8,9,10,11,12,13,14,15);
}

// B-operand fragment (32x16 f16), B(k,n) = W[n0+n][k] with W row-major in LDS.
// lanes 0-15 column N=lane hold K=k0..k0+7, k0+16..23; lanes 16-31 K=k0+8..15, k0+24..31.
__device__ __forceinline__ v16h load_frag_B(const _Float16* base, int stride,
                                            int n0, int k0, int lane) {
  int n    = lane & 15;
  int koff = k0 + ((lane >> 4) << 3);  // 0 or 8
  const _Float16* p = base + (n0 + n) * stride + koff;
  v8h lo = *(const v8h*)(p);
  v8h hi = *(const v8h*)(p + 16);
  return __builtin_shufflevector(lo, hi, 0,1,2,3,4,5,6,7,8,9,10,11,12,13,14,15);
}

// CDNA5 hardware tanh (TRANS32 op).
__device__ __forceinline__ float hw_tanh(float x) {
#if __has_builtin(__builtin_amdgcn_tanhf)
  return __builtin_amdgcn_tanhf(x);
#elif __has_builtin(__builtin_amdgcn_tanh_f32)
  return __builtin_amdgcn_tanh_f32(x);
#else
  float r;
  asm("v_tanh_f32 %0, %1\n\tv_nop" : "=v"(r) : "v"(x));
  return r;
#endif
}
// sigmoid(x) = 0.5*tanh(x/2) + 0.5 : one trans op + FMA, no IEEE divide.
__device__ __forceinline__ float hw_sigmoid(float x) {
  return __builtin_fmaf(0.5f, hw_tanh(0.5f * x), 0.5f);
}

__global__ void __launch_bounds__(256, 1)
prnn_gru_wmma_kernel(const float* __restrict__ x,
                     const float* __restrict__ w_ih,
                     const float* __restrict__ w_hh,
                     const float* __restrict__ b_ih,
                     const float* __restrict__ b_hh,
                     const float* __restrict__ w1,
                     const float* __restrict__ b1,
                     const float* __restrict__ w2,
                     const float* __restrict__ b2,
                     float* __restrict__ out)
{
  extern __shared__ char smem[];
  _Float16* s_whh = (_Float16*)smem;                 // [384][HS] f16, gates (r,z,n)
  _Float16* s_w1  = s_whh + WHH_HALVES;              // [128][HS] f16
  _Float16* s_w2  = s_w1  + W1_HALVES;               // [16][HS]  f16, rows 4..15 = 0
  float*    s_lut  = (float*)(s_w2 + W2_HALVES);     // [4][384]
  float*    s_bhhn = s_lut + LUT_FLOATS;             // [128] b_hh for n gate
  float*    s_b1   = s_bhhn + NH;                    // [128]
  float*    s_b2   = s_b1 + NH;                      // [16], cols 4..15 = 0
  unsigned* s_bits = (unsigned*)(s_b2 + 16);         // [128 rows][2 words]
  _Float16* s_hA   = (_Float16*)(s_bits + ROWS_PER_BLOCK * 2);
  _Float16* s_hB   = s_hA + HBUF_HALVES;
  _Float16* s_hidb = s_hB + HBUF_HALVES;
  float*    s_xstage = (float*)s_hidb;               // 32 KB x tile (reused as hid later)

  const int tid     = threadIdx.x;       // 0..255
  const int lane    = tid & 31;
  const int wave    = tid >> 5;          // 0..7, each owns 16 batch rows
  const int rowBase = blockIdx.x * ROWS_PER_BLOCK;

  // ---- TDM: async-stage the contiguous 128x64 f32 x tile into LDS ---------
  if (wave == 0) {
    unsigned lds_off = (unsigned)(unsigned long long)(uintptr_t)s_xstage;
    unsigned long long ga =
        (unsigned long long)(uintptr_t)(x + (size_t)rowBase * 64);
    const unsigned td0 = 128u * 64u;           // 1-D tensor: 8192 f32 elements
    u32x4 g0 = { 1u,                           // count=1 (valid user D#)
                 lds_off,                      // lds_addr
                 (unsigned)(ga & 0xffffffffu), // global_addr[31:0]
                 (unsigned)((ga >> 32) & 0x01ffffffu) | (2u << 30) }; // [56:32] | type=2
    i32x8 g1 = { (int)(2u << 16),              // workgroup_mask=0, data_size=2 (4 B)
                 (int)((td0 & 0xffffu) << 16), // tensor_dim0[15:0]
                 (int)(td0 >> 16) | (1 << 16), // tensor_dim0[31:16], tensor_dim1=1
                 (int)((td0 & 0xffffu) << 16), // tile_dim0 = 8192
                 0,                            // tile_dim1=0 (1-D), tile_dim2=0
                 (int)td0,                     // tensor_dim0_stride[31:0]
                 0, 0 };
    i32x4 gz4 = { 0, 0, 0, 0 };
    i32x8 gz8 = { 0, 0, 0, 0, 0, 0, 0, 0 };
    __builtin_amdgcn_tensor_load_to_lds(g0, g1, gz4, gz4, gz8, 0);
  }

  if (tid == 0) {                        // warm L2 for the weight broadcast
    __builtin_prefetch(w_hh, 0, 1);
    __builtin_prefetch(w1, 0, 1);
  }

  // ---- stage weights / LUTs into LDS (overlaps with the TDM transfer) -----
  for (int j = tid; j < NGATE * NH; j += blockDim.x) {
    int r = j >> 7, c = j & 127;
    s_whh[r * HS + c] = (_Float16)w_hh[j];
  }
  for (int j = tid; j < NH * NH; j += blockDim.x) {
    int r = j >> 7, c = j & 127;
    s_w1[r * HS + c] = (_Float16)w1[j];
  }
  for (int j = tid; j < 16 * NH; j += blockDim.x) {
    int r = j >> 7, c = j & 127;
    s_w2[r * HS + c] = (r < 4) ? (_Float16)w2[r * NH + c] : (_Float16)0.0f;
  }
  // xi LUT: only 4 possible input projections per column since P==2.
  for (int j = tid; j < 4 * NGATE; j += blockDim.x) {
    int d = j / NGATE, g = j - d * NGATE;
    float v = b_ih[g];
    if (g < 2 * NH) v += b_hh[g];        // fold b_hh into r,z pre-activations
    if (d & 1) v += w_ih[g * 2 + 0];
    if (d & 2) v += w_ih[g * 2 + 1];
    s_lut[j] = v;
  }
  for (int j = tid; j < NH; j += blockDim.x) {
    s_bhhn[j] = b_hh[2 * NH + j];
    s_b1[j]   = b1[j];
  }
  if (tid < 16) s_b2[tid] = (tid < 4) ? b2[tid] : 0.0f;
  for (int j = tid; j < 2 * HBUF_HALVES; j += blockDim.x) s_hA[j] = (_Float16)0.0f;

  if (wave == 0) __builtin_amdgcn_s_wait_tensorcnt(0);   // x tile landed
  __syncthreads();

  // ---- pack x bit-streams from the LDS-staged tile ------------------------
  {
    int row = tid >> 1, half = tid & 1;  // 256 threads -> 128 rows x 2 words
    const float* xp = s_xstage + row * 64 + half * 32;
    unsigned m = 0u;
#pragma unroll
    for (int i = 0; i < 32; ++i) m |= (xp[i] > 0.5f) ? (1u << i) : 0u;
    s_bits[row * 2 + half] = m;
  }
  __syncthreads();                       // also fences x-stage before hid reuse

  // ---- per-wave recurrence -------------------------------------------------
  const int lhalf = lane >> 4;           // 0 / 1
  const int lcol  = lane & 15;

  _Float16* hcur  = s_hA + wave * 16 * HS;
  _Float16* hnext = s_hB + wave * 16 * HS;
  _Float16* hid   = s_hidb + wave * 16 * HS;

  // loop-invariant head fragments: padded w2^T and b2 broadcast
  v16h bw2[4];
#pragma unroll
  for (int k = 0; k < 4; ++k) bw2[k] = load_frag_B(s_w2, HS, 0, k * 32, lane);
  const float b2v = s_b2[lcol];

  // hoist per-row patch streams into registers (8 rows per lane-half)
  unsigned long long bb[8];
#pragma unroll
  for (int i = 0; i < 8; ++i) {
    int row = wave * 16 + i + 8 * lhalf;
    bb[i] = ((unsigned long long)s_bits[row * 2 + 1] << 32) | s_bits[row * 2 + 0];
  }

  float lp[8] = {0.f, 0.f, 0.f, 0.f, 0.f, 0.f, 0.f, 0.f};

  for (int t = 0; t < T_STEPS; ++t) {
    int din[8], idx[8];
    const int bidx = 2 * t;
#pragma unroll
    for (int i = 0; i < 8; ++i) {
      idx[i] = (int)((bb[i] >> bidx) & 3ull);
      din[i] = (t == 0) ? 0 : (int)((bb[i] >> (bidx - 2)) & 3ull);
    }

    // A-fragments of current hidden state (16 rows x K=128)
    v16h ah[4];
#pragma unroll
    for (int k = 0; k < 4; ++k) ah[k] = load_frag_A(hcur, HS, k * 32, lane);

    // ---- GRU gates, 16-column hidden chunks: 12 WMMA each ----
#pragma unroll
    for (int n = 0; n < 8; ++n) {
      v8f accr = {}; v8f accz = {};
      const float bn = s_bhhn[n * 16 + lcol];
      v8f accn = {bn, bn, bn, bn, bn, bn, bn, bn};
#pragma unroll
      for (int k = 0; k < 4; ++k) {
        v16h br  = load_frag_B(s_whh, HS, 0 * NH + n * 16, k * 32, lane);
        v16h bz  = load_frag_B(s_whh, HS, 1 * NH + n * 16, k * 32, lane);
        v16h bnn = load_frag_B(s_whh, HS, 2 * NH + n * 16, k * 32, lane);
        accr = wmma_f16(ah[k], br,  accr);
        accz = wmma_f16(ah[k], bz,  accz);
        accn = wmma_f16(ah[k], bnn, accn);
      }
#pragma unroll
      for (int i = 0; i < 8; ++i) {
        const int    col  = n * 16 + lcol;
        const float* lutd = s_lut + din[i] * NGATE;
        float rg = hw_sigmoid(lutd[col] + accr[i]);
        float zg = hw_sigmoid(lutd[NH + col] + accz[i]);
        float ng = hw_tanh(lutd[2 * NH + col] + rg * accn[i]);
        const int row = i + 8 * lhalf;
        float hold = (float)hcur[row * HS + col];
        hnext[row * HS + col] = (_Float16)((1.0f - zg) * ng + zg * hold);
      }
    }
    { _Float16* tmp = hcur; hcur = hnext; hnext = tmp; }   // h <- h_new

    // ---- head: hid^T = relu(w1 @ h^T + b1), A = w1 tile, B = h rows ----
    v16h bh[4];
#pragma unroll
    for (int k = 0; k < 4; ++k) bh[k] = load_frag_B(hcur, HS, 0, k * 32, lane);
#pragma unroll
    for (int m = 0; m < 8; ++m) {
      v8f acc;
#pragma unroll
      for (int i = 0; i < 8; ++i) acc[i] = s_b1[m * 16 + i + 8 * lhalf];
#pragma unroll
      for (int k = 0; k < 4; ++k) {
        v16h aw = load_frag_A(s_w1 + m * 16 * HS, HS, k * 32, lane);
        acc = wmma_f16(aw, bh[k], acc);
      }
#pragma unroll
      for (int i = 0; i < 8; ++i) {
        float v = acc[i] > 0.0f ? acc[i] : 0.0f;            // ReLU
        hid[lcol * HS + m * 16 + i + 8 * lhalf] = (_Float16)v;  // hid[row][feat]
      }
    }

    // ---- logits = hid @ w2^T + b2 (cols 0..3 valid), then log-softmax ----
    v8f acc2 = {b2v, b2v, b2v, b2v, b2v, b2v, b2v, b2v};
#pragma unroll
    for (int k = 0; k < 4; ++k) {
      v16h ahid = load_frag_A(hid, HS, k * 32, lane);
      acc2 = wmma_f16(ahid, bw2[k], acc2);
    }
#pragma unroll
    for (int i = 0; i < 8; ++i) {
      float v  = acc2[i];
      float m1 = fmaxf(v,  __shfl_xor(v,  1, 32));
      float mx = fmaxf(m1, __shfl_xor(m1, 2, 32));          // max over 4 gates
      float e  = __expf(v - mx);
      float s1 = e + __shfl_xor(e, 1, 32);
      float ss = s1 + __shfl_xor(s1, 2, 32);                // sum over 4 gates
      float lpg = (v - mx) - __logf(ss);                    // log softmax, this gate
      float sel = __shfl(lpg, idx[i] + (lhalf << 4), 32);   // pick observed patch
      lp[i] += sel;
    }
  }

  // ---- write per-row log-prob sums (lanes 0 and 16 hold row r / r+8) -------
  if (lcol == 0) {
#pragma unroll
    for (int i = 0; i < 8; ++i) {
      out[rowBase + wave * 16 + i + 8 * lhalf] = lp[i];
    }
  }
}

extern "C" void kernel_launch(void* const* d_in, const int* in_sizes, int n_in,
                              void* d_out, int out_size, void* d_ws, size_t ws_size,
                              hipStream_t stream) {
  const float* x    = (const float*)d_in[0];
  const float* w_ih = (const float*)d_in[1];
  const float* w_hh = (const float*)d_in[2];
  const float* b_ih = (const float*)d_in[3];
  const float* b_hh = (const float*)d_in[4];
  const float* w1   = (const float*)d_in[5];
  const float* b1   = (const float*)d_in[6];
  const float* w2   = (const float*)d_in[7];
  const float* b2   = (const float*)d_in[8];
  float* out = (float*)d_out;

  dim3 grid(BATCH / ROWS_PER_BLOCK);   // 128 blocks
  dim3 block(256);                     // 8 waves, 2 per SIMD32
  prnn_gru_wmma_kernel<<<grid, block, SMEM_BYTES, stream>>>(
      x, w_ih, w_hh, b_ih, b_hh, w1, b1, w2, b2, out);
}
